// MWAFM_Net_84456236908920
// MI455X (gfx1250) — matrix operand order repaired and checked by reading, not compiled, not run on hardware
//
#include <hip/hip_runtime.h>
#include <cmath>

// ---------------------------------------------------------------------------
// Types
// ---------------------------------------------------------------------------
typedef __attribute__((ext_vector_type(16))) _Float16 v16h;
typedef __attribute__((ext_vector_type(8)))  _Float16 v8h;
typedef __attribute__((ext_vector_type(8)))  float    v8f;

#define TILE_M 128
#define TILE_N 128
#define TILE_K 32
#define LDSS   40   // halves per LDS row: 80 bytes, keeps 16B alignment per row

// Convert 16 consecutive f32 (16B aligned) -> 16 f16, store as 2x16B to LDS.
__device__ __forceinline__ void cvt16_store(const float* __restrict__ src,
                                            _Float16* dst)
{
  const float4 f0 = *(const float4*)(src);
  const float4 f1 = *(const float4*)(src + 4);
  const float4 f2 = *(const float4*)(src + 8);
  const float4 f3 = *(const float4*)(src + 12);
  v8h h0, h1;
  h0[0] = (_Float16)f0.x; h0[1] = (_Float16)f0.y; h0[2] = (_Float16)f0.z; h0[3] = (_Float16)f0.w;
  h0[4] = (_Float16)f1.x; h0[5] = (_Float16)f1.y; h0[6] = (_Float16)f1.z; h0[7] = (_Float16)f1.w;
  h1[0] = (_Float16)f2.x; h1[1] = (_Float16)f2.y; h1[2] = (_Float16)f2.z; h1[3] = (_Float16)f2.w;
  h1[4] = (_Float16)f3.x; h1[5] = (_Float16)f3.y; h1[6] = (_Float16)f3.z; h1[7] = (_Float16)f3.w;
  *(v8h*)dst       = h0;
  *(v8h*)(dst + 8) = h1;
}

// Gather 16 f32 strided by `stride`, convert, store contiguously (2x16B) to LDS.
__device__ __forceinline__ void cvt16_store_gather(const float* __restrict__ base,
                                                   long stride, _Float16* dst)
{
  float f[16];
  #pragma unroll
  for (int j = 0; j < 16; ++j) f[j] = base[(long)j * stride];
  v8h h0, h1;
  #pragma unroll
  for (int j = 0; j < 8; ++j) { h0[j] = (_Float16)f[j]; h1[j] = (_Float16)f[j + 8]; }
  *(v8h*)dst       = h0;
  *(v8h*)(dst + 8) = h1;
}

// ---------------------------------------------------------------------------
// WMMA GEMM:  C = act(alpha * A @ B (+bias))   (f32 storage, f16 WMMA core)
//   A: [M,K] row-major, lda
//   B: NN -> [K,N] ldb ; TRANS_B -> [N,K] ldb
//   batched: grid.z = batch*H ; zb = z/H, zh = z%H with separate strides
// ---------------------------------------------------------------------------
template <bool TRANS_B>
__global__ __launch_bounds__(256) void gemm_wmma_kernel(
    const float* __restrict__ A, const float* __restrict__ Bm,
    const float* __restrict__ bias, float* __restrict__ C,
    int M, int N, int K,
    long lda, long ldb, long ldc,
    int H, long sAb, long sAh, long sBb, long sBh, long sCb, long sCh,
    float alpha, int act)
{
  __shared__ __align__(16) _Float16 As[TILE_M][LDSS];  // As[m][k]
  __shared__ __align__(16) _Float16 Bs[TILE_N][LDSS];  // Bs[n][k]

  const int z  = blockIdx.z;
  const int zb = z / H, zh = z % H;
  A  += (long)zb * sAb + (long)zh * sAh;
  Bm += (long)zb * sBb + (long)zh * sBh;
  C  += (long)zb * sCb + (long)zh * sCh;

  const int n0   = blockIdx.x * TILE_N;
  const int m0   = blockIdx.y * TILE_M;
  const int tid  = threadIdx.x;
  const int lane = tid & 31;
  const int wave = tid >> 5;
  const int wm0  = (wave & 3) * 32;   // 4 waves along M (32 rows each)
  const int wn0  = (wave >> 2) * 64;  // 2 waves along N (64 cols each)

  v8f acc[2][4];
  for (int i = 0; i < 2; ++i)
    for (int j = 0; j < 4; ++j)
      for (int e = 0; e < 8; ++e) acc[i][j][e] = 0.0f;

  const int arow = tid >> 1;        // 0..127
  const int akk  = (tid & 1) * 16;  // 0 or 16

  const bool fullM = (m0 + TILE_M <= M);
  const bool fullN = (n0 + TILE_N <= N);
  const bool ldaOK = ((lda & 3) == 0);
  const bool ldbOK = ((ldb & 3) == 0);

  for (int k0 = 0; k0 < K; k0 += TILE_K) {
    const bool fullK = (k0 + TILE_K <= K);

    // ---- stage A tile (f32 -> f16) ----
    {
      const int gm = m0 + arow;
      if (fullK && gm < M)
        __builtin_prefetch(&A[(long)gm * lda + (k0 + TILE_K)], 0, 1);
      if (fullM && fullK && ldaOK) {
        cvt16_store(&A[(long)gm * lda + k0 + akk], &As[arow][akk]);
      } else if (gm < M) {
        #pragma unroll
        for (int j = 0; j < 16; ++j) {
          const int gk = k0 + akk + j;
          As[arow][akk + j] = (_Float16)((gk < K) ? A[(long)gm * lda + gk] : 0.0f);
        }
      } else {
        #pragma unroll
        for (int j = 0; j < 16; ++j) As[arow][akk + j] = (_Float16)0.0f;
      }
    }
    // ---- stage B tile into Bs[n][k] ----
    if (TRANS_B) {
      const int gn = n0 + arow;
      if (fullK && gn < N)
        __builtin_prefetch(&Bm[(long)gn * ldb + (k0 + TILE_K)], 0, 1);
      if (fullN && fullK && ldbOK) {
        cvt16_store(&Bm[(long)gn * ldb + k0 + akk], &Bs[arow][akk]);
      } else if (gn < N) {
        #pragma unroll
        for (int j = 0; j < 16; ++j) {
          const int gk = k0 + akk + j;
          Bs[arow][akk + j] = (_Float16)((gk < K) ? Bm[(long)gn * ldb + gk] : 0.0f);
        }
      } else {
        #pragma unroll
        for (int j = 0; j < 16; ++j) Bs[arow][akk + j] = (_Float16)0.0f;
      }
    } else {
      if (fullN && fullK) {
        // column gather: thread covers column n0+arow, K chunk akk..akk+15
        cvt16_store_gather(&Bm[(long)(k0 + akk) * ldb + n0 + arow], ldb, &Bs[arow][akk]);
      } else {
        const int bk  = tid >> 3;         // 0..31
        const int bnn = (tid & 7) * 16;   // 0..112
        const int gk  = k0 + bk;
        if (gk < K) {
          #pragma unroll
          for (int j = 0; j < 16; ++j) {
            const int gn = n0 + bnn + j;
            Bs[bnn + j][bk] = (_Float16)((gn < N) ? Bm[(long)gk * ldb + gn] : 0.0f);
          }
        } else {
          #pragma unroll
          for (int j = 0; j < 16; ++j) Bs[bnn + j][bk] = (_Float16)0.0f;
        }
      }
    }
    __syncthreads();

    // ---- fragments (ISA 16-bit A/B wave32 layout) ----
    {
      const int rsel = lane & 15;
      const int hi   = lane >> 4;
      v16h afrag[2], bfrag[4];
      #pragma unroll
      for (int i = 0; i < 2; ++i) {
        const _Float16* ap = &As[wm0 + i * 16 + rsel][0];
        const v8h l0 = *(const v8h*)(ap + hi * 8);        // K = hi*8 .. +7
        const v8h l1 = *(const v8h*)(ap + hi * 8 + 16);   // K = hi*8+16 .. +7
        #pragma unroll
        for (int e = 0; e < 8; ++e) { afrag[i][e] = l0[e]; afrag[i][e + 8] = l1[e]; }
      }
      #pragma unroll
      for (int j = 0; j < 4; ++j) {
        const _Float16* bp = &Bs[wn0 + j * 16 + rsel][0];
        const v8h l0 = *(const v8h*)(bp + hi * 8);        // K = hi*8 .. +7
        const v8h l1 = *(const v8h*)(bp + hi * 8 + 16);   // K = hi*8+16 .. +7
        #pragma unroll
        for (int e = 0; e < 8; ++e) { bfrag[j][e] = l0[e]; bfrag[j][e + 8] = l1[e]; }
      }
      #pragma unroll
      for (int i = 0; i < 2; ++i)
        #pragma unroll
        for (int j = 0; j < 4; ++j)
          acc[i][j] = __builtin_amdgcn_wmma_f32_16x16x32_f16(
              false, afrag[i], false, bfrag[j], (short)0, acc[i][j], false, false);
    }
    __syncthreads();
  }

  // ---- epilogue ----
  const int ecol = lane & 15;
  const int ehi  = lane >> 4;
  if (fullM && fullN) {
    #pragma unroll
    for (int i = 0; i < 2; ++i) {
      #pragma unroll
      for (int j = 0; j < 4; ++j) {
        const int gn = n0 + wn0 + j * 16 + ecol;
        const float bv = bias ? bias[gn] : 0.0f;
        #pragma unroll
        for (int r = 0; r < 8; ++r) {
          const int gm = m0 + wm0 + i * 16 + r + ehi * 8;
          float v = acc[i][j][r] * alpha + bv;
          if (act == 1)      v = fmaxf(v, 0.0f);
          else if (act == 2) v = tanhf(v);
          C[(long)gm * ldc + gn] = v;
        }
      }
    }
  } else {
    #pragma unroll
    for (int i = 0; i < 2; ++i) {
      #pragma unroll
      for (int j = 0; j < 4; ++j) {
        #pragma unroll
        for (int r = 0; r < 8; ++r) {
          const int gm = m0 + wm0 + i * 16 + r + ehi * 8;
          const int gn = n0 + wn0 + j * 16 + ecol;
          if (gm < M && gn < N) {
            float v = acc[i][j][r] * alpha;
            if (bias) v += bias[gn];
            if (act == 1)      v = fmaxf(v, 0.0f);
            else if (act == 2) v = tanhf(v);
            C[(long)gm * ldc + gn] = v;
          }
        }
      }
    }
  }
}

// ---------------------------------------------------------------------------
// Softmax over rows (wave32 per row)
// ---------------------------------------------------------------------------
__global__ __launch_bounds__(256) void softmax_rows_kernel(
    float* __restrict__ x, long rows, int n)
{
  const long row = (long)blockIdx.x * 8 + (threadIdx.x >> 5);
  if (row >= rows) return;
  const int lane = threadIdx.x & 31;
  float* p = x + row * (long)n;
  float mx = -1e30f;
  for (int i = lane; i < n; i += 32) mx = fmaxf(mx, p[i]);
  for (int s = 16; s > 0; s >>= 1) mx = fmaxf(mx, __shfl_xor(mx, s, 32));
  float sum = 0.0f;
  for (int i = lane; i < n; i += 32) { const float e = __expf(p[i] - mx); p[i] = e; sum += e; }
  for (int s = 16; s > 0; s >>= 1) sum += __shfl_xor(sum, s, 32);
  const float inv = 1.0f / sum;
  for (int i = lane; i < n; i += 32) p[i] *= inv;
}

// ---------------------------------------------------------------------------
// LayerNorm over D=512: out = LN(a [+b] [+c]) * gamma + beta  (accum: out+=)
// gb: [gamma(512); beta(512)]
// ---------------------------------------------------------------------------
__global__ __launch_bounds__(256) void ln512_kernel(
    const float* __restrict__ a, const float* __restrict__ b,
    const float* __restrict__ c, const float* __restrict__ gb,
    float* __restrict__ out, int accum)
{
  const long row = blockIdx.x;
  const int  t   = threadIdx.x;
  const long o0  = row * 512 + t, o1 = o0 + 256;
  float x0 = a[o0], x1 = a[o1];
  if (b) { x0 += b[o0]; x1 += b[o1]; }
  if (c) { x0 += c[o0]; x1 += c[o1]; }

  __shared__ float red[8];
  __shared__ float stat[2];
  const int lane = t & 31, wid = t >> 5;

  float s = x0 + x1;
  for (int sh = 16; sh > 0; sh >>= 1) s += __shfl_xor(s, sh, 32);
  if (lane == 0) red[wid] = s;
  __syncthreads();
  if (t == 0) { float tt = 0; for (int i = 0; i < 8; ++i) tt += red[i]; stat[0] = tt * (1.0f / 512.0f); }
  __syncthreads();
  const float m  = stat[0];
  const float d0 = x0 - m, d1 = x1 - m;
  float vv = d0 * d0 + d1 * d1;
  for (int sh = 16; sh > 0; sh >>= 1) vv += __shfl_xor(vv, sh, 32);
  if (lane == 0) red[wid] = vv;
  __syncthreads();
  if (t == 0) { float tt = 0; for (int i = 0; i < 8; ++i) tt += red[i]; stat[1] = rsqrtf(tt * (1.0f / 512.0f) + 1e-5f); }
  __syncthreads();
  const float r  = stat[1];
  const float y0 = d0 * r * gb[t]       + gb[512 + t];
  const float y1 = d1 * r * gb[t + 256] + gb[768 + t];
  if (accum) { out[o0] += y0; out[o1] += y1; }
  else       { out[o0]  = y0; out[o1]  = y1; }
}

// ---------------------------------------------------------------------------
// Sliding-window (banded) attention: H=4, hd=128, wave per (b,t,h)
// ---------------------------------------------------------------------------
template <int W>
__global__ __launch_bounds__(256) void sliding_attn_kernel(
    const float* __restrict__ q, const float* __restrict__ k,
    const float* __restrict__ v, float* __restrict__ out,
    int Bn, int Tn, float scale)
{
  const int H = 4, HD = 128, D = 512;
  const int wave = blockIdx.x * (blockDim.x >> 5) + (threadIdx.x >> 5);
  if (wave >= Bn * Tn * H) return;
  const int lane = threadIdx.x & 31;
  const int h = wave % H;
  const int t = (wave / H) % Tn;
  const int b = wave / (H * Tn);
  const long base = ((long)(b * Tn + t)) * D + h * HD + lane * 4;
  const float4 q4 = *(const float4*)(q + base);

  float sc[2 * W + 1];
  #pragma unroll
  for (int j = 0; j < 2 * W + 1; ++j) {
    const int tt = t + j - W;
    float s = -1e30f;
    if (tt >= 0 && tt < Tn) {
      const long kb = ((long)(b * Tn + tt)) * D + h * HD + lane * 4;
      const float4 k4 = *(const float4*)(k + kb);
      float d = q4.x * k4.x + q4.y * k4.y + q4.z * k4.z + q4.w * k4.w;
      #pragma unroll
      for (int sh = 16; sh > 0; sh >>= 1) d += __shfl_xor(d, sh, 32);
      s = d * scale;
    }
    sc[j] = s;
  }
  float mx = -1e30f;
  #pragma unroll
  for (int j = 0; j < 2 * W + 1; ++j) mx = fmaxf(mx, sc[j]);
  float sum = 0.0f;
  #pragma unroll
  for (int j = 0; j < 2 * W + 1; ++j) { sc[j] = __expf(sc[j] - mx); sum += sc[j]; }
  const float inv = 1.0f / sum;

  float4 o4 = make_float4(0.f, 0.f, 0.f, 0.f);
  #pragma unroll
  for (int j = 0; j < 2 * W + 1; ++j) {
    const int tt = t + j - W;
    if (tt >= 0 && tt < Tn) {
      const long vb = ((long)(b * Tn + tt)) * D + h * HD + lane * 4;
      const float4 v4 = *(const float4*)(v + vb);
      const float p = sc[j] * inv;
      o4.x += p * v4.x; o4.y += p * v4.y; o4.z += p * v4.z; o4.w += p * v4.w;
    }
  }
  *(float4*)(out + base) = o4;
}

// ---------------------------------------------------------------------------
// mean over axis 1: out[b, oOff+d] = mean_t x[b,t,d]
// ---------------------------------------------------------------------------
__global__ void mean_axis1_kernel(const float* __restrict__ x, float* __restrict__ out,
                                  int Bn, int Tn, int Dd, int oLd, int oOff)
{
  const int i = blockIdx.x * blockDim.x + threadIdx.x;
  if (i >= Bn * Dd) return;
  const int b = i / Dd, d = i % Dd;
  const float* p = x + (long)b * Tn * Dd + d;
  float s = 0.0f;
  for (int t = 0; t < Tn; ++t) s += p[(long)t * Dd];
  out[(long)b * oLd + oOff + d] = s * (1.0f / (float)Tn);
}

// ---------------------------------------------------------------------------
// Host orchestration
// ---------------------------------------------------------------------------
static void launch_gemm(hipStream_t st, const float* A, const float* Bm, const float* bias,
                        float* C, int M, int N, int K, long lda, long ldb, long ldc,
                        int batch, int H, long sAb, long sAh, long sBb, long sBh,
                        long sCb, long sCh, float alpha, int act, bool transB)
{
  dim3 grid((N + TILE_N - 1) / TILE_N, (M + TILE_M - 1) / TILE_M, batch * H);
  if (transB)
    gemm_wmma_kernel<true><<<grid, 256, 0, st>>>(A, Bm, bias, C, M, N, K, lda, ldb, ldc,
                                                 H, sAb, sAh, sBb, sBh, sCb, sCh, alpha, act);
  else
    gemm_wmma_kernel<false><<<grid, 256, 0, st>>>(A, Bm, bias, C, M, N, K, lda, ldb, ldc,
                                                  H, sAb, sAh, sBb, sBh, sCb, sCh, alpha, act);
}

// torch-style MHA: qsrc:[B,Lq,512], kvsrc:[B,Lk,512] -> outp:[B,Lq,512]
static void run_mha(hipStream_t st, const float* qsrc, int Lq, const float* kvsrc, int Lk,
                    const float* in_w, const float* in_b, long in_ld,
                    const float* out_w, const float* out_b, int h,
                    float* qp, float* kp, float* vp, float* scores, float* obuf, float* outp)
{
  const int Dd = 512, Bn = 32;
  const int hd = Dd / h;
  launch_gemm(st, qsrc,  in_w,          in_b,          qp, Bn * Lq, Dd, Dd, Dd, in_ld, Dd,
              1, 1, 0, 0, 0, 0, 0, 0, 1.0f, 0, false);
  launch_gemm(st, kvsrc, in_w + Dd,     in_b + Dd,     kp, Bn * Lk, Dd, Dd, Dd, in_ld, Dd,
              1, 1, 0, 0, 0, 0, 0, 0, 1.0f, 0, false);
  launch_gemm(st, kvsrc, in_w + 2 * Dd, in_b + 2 * Dd, vp, Bn * Lk, Dd, Dd, Dd, in_ld, Dd,
              1, 1, 0, 0, 0, 0, 0, 0, 1.0f, 0, false);
  const float scale = 1.0f / sqrtf((float)hd);
  // scores[b,h] = qp[b,:,h] @ kp[b,:,h]^T
  launch_gemm(st, qp, kp, nullptr, scores, Lq, Lk, hd, Dd, Dd, Lk,
              Bn, h, (long)Lq * Dd, hd, (long)Lk * Dd, hd,
              (long)h * Lq * Lk, (long)Lq * Lk, scale, 0, true);
  const long rows = (long)Bn * h * Lq;
  softmax_rows_kernel<<<dim3((unsigned)((rows + 7) / 8)), 256, 0, st>>>(scores, rows, Lk);
  // obuf[b,:,h] = P @ vp
  launch_gemm(st, scores, vp, nullptr, obuf, Lq, hd, Lk, Lk, Dd, Dd,
              Bn, h, (long)h * Lq * Lk, (long)Lq * Lk, (long)Lk * Dd, hd,
              (long)Lq * Dd, hd, 1.0f, 0, false);
  launch_gemm(st, obuf, out_w, out_b, outp, Bn * Lq, Dd, Dd, Dd, Dd, Dd,
              1, 1, 0, 0, 0, 0, 0, 0, 1.0f, 0, false);
}

static void run_qq_block(hipStream_t st, const float* att, int rows,
                         const float* l1w, const float* l1b,
                         const float* l2w, const float* l2b,
                         const float* lnp, float* t1, float* t2, float* outp)
{
  launch_gemm(st, att, l1w, l1b, t1, rows, 512, 512, 512, 512, 512,
              1, 1, 0, 0, 0, 0, 0, 0, 1.0f, 1, false);  // ReLU
  launch_gemm(st, t1, l2w, l2b, t2, rows, 512, 512, 512, 512, 512,
              1, 1, 0, 0, 0, 0, 0, 0, 1.0f, 0, false);
  ln512_kernel<<<rows, 256, 0, st>>>(att, t2, nullptr, lnp, outp, 0);
}

extern "C" void kernel_launch(void* const* d_in, const int* in_sizes, int n_in,
                              void* d_out, int out_size, void* d_ws, size_t ws_size,
                              hipStream_t stream)
{
  (void)in_sizes; (void)n_in; (void)out_size; (void)ws_size;
  const int Bn = 32, T = 480, LQ = 20, Dd = 512;

  const float* audio       = (const float*)d_in[0];
  const float* question    = (const float*)d_in[1];
  const float* audio_fc_w  = (const float*)d_in[2];
  const float* audio_fc_b  = (const float*)d_in[3];
  const float* qst_fc_w    = (const float*)d_in[4];
  const float* qst_fc_b    = (const float*)d_in[5];
  const float* attnq_in_w  = (const float*)d_in[6];
  const float* attnq_in_b  = (const float*)d_in[7];
  const float* attnq_out_w = (const float*)d_in[8];
  const float* attnq_out_b = (const float*)d_in[9];
  const float* qq_lin1_w   = (const float*)d_in[10];
  const float* qq_lin1_b   = (const float*)d_in[11];
  const float* qq_lin2_w   = (const float*)d_in[12];
  const float* qq_lin2_b   = (const float*)d_in[13];
  const float* qq_ln       = (const float*)d_in[14];
  const float* sa_in_w     = (const float*)d_in[15];
  const float* sa_in_b     = (const float*)d_in[16];
  const float* sa_out_w    = (const float*)d_in[17];
  const float* sa_out_b    = (const float*)d_in[18];
  const float* cm_in_w     = (const float*)d_in[19];
  const float* cm_in_b     = (const float*)d_in[20];
  const float* cm_out_w    = (const float*)d_in[21];
  const float* cm_out_b    = (const float*)d_in[22];
  const float* ff1_w       = (const float*)d_in[23];
  const float* ff1_b       = (const float*)d_in[24];
  const float* ff2_w       = (const float*)d_in[25];
  const float* ff2_b       = (const float*)d_in[26];
  const float* ln1         = (const float*)d_in[27];
  const float* ln2         = (const float*)d_in[28];
  const float* ms_qkv_w    = (const float*)d_in[29];
  const float* ms_qkv_b    = (const float*)d_in[30];
  const float* ms_ln       = (const float*)d_in[31];
  const float* msl_w       = (const float*)d_in[32];
  const float* msl_b       = (const float*)d_in[33];
  const float* ms_norm     = (const float*)d_in[34];
  const float* qa_in_w     = (const float*)d_in[35];
  const float* qa_in_b     = (const float*)d_in[36];
  const float* qa_out_w    = (const float*)d_in[37];
  const float* qa_out_b    = (const float*)d_in[38];
  const float* qa_ff1_w    = (const float*)d_in[39];
  const float* qa_ff1_b    = (const float*)d_in[40];
  const float* qa_ff2_w    = (const float*)d_in[41];
  const float* qa_ff2_b    = (const float*)d_in[42];
  const float* qa_ln1      = (const float*)d_in[43];
  const float* qa_ln2      = (const float*)d_in[44];
  const float* fc1_w       = (const float*)d_in[45];
  const float* fc1_b       = (const float*)d_in[46];
  const float* fc2_w       = (const float*)d_in[47];
  const float* fc2_b       = (const float*)d_in[48];
  const float* pred_w      = (const float*)d_in[49];
  const float* pred_b      = (const float*)d_in[50];

  // ---- workspace bump allocator ----
  size_t off = 0;
  auto alloc = [&](size_t nf) -> float* {
    float* p = (float*)((char*)d_ws + off);
    off += ((nf * sizeof(float) + 255) & ~(size_t)255);
    return p;
  };
  const size_t big = (size_t)Bn * T * Dd;   // 7.86M floats
  const size_t sm  = (size_t)Bn * LQ * Dd;  // 327K floats
  float* AF   = alloc(big);
  float* QF   = alloc(sm);
  float* QPs  = alloc(sm);
  float* KPs  = alloc(sm);
  float* VPs  = alloc(sm);
  float* SCq  = alloc((size_t)Bn * 4 * LQ * LQ);
  float* AOq  = alloc(sm);
  float* APq  = alloc(sm);
  float* T1s  = alloc(sm);
  float* T2s  = alloc(sm);
  float* QSTF = alloc(sm);
  float* G1   = alloc(big);
  float* G2   = alloc(big);
  float* G3   = alloc(big);
  float* O1   = alloc(big);
  float* R1   = alloc(big);
  float* R2   = alloc(big);
  float* SC   = alloc((size_t)Bn * T * T);
  float* COMB = alloc((size_t)Bn * 1024);
  float* C1   = alloc((size_t)Bn * 512);
  float* C2   = alloc((size_t)Bn * 256);

  // 1) af = audio @ audio_fc_w + b
  launch_gemm(stream, audio, audio_fc_w, audio_fc_b, AF, Bn * T, Dd, 128, 128, Dd, Dd,
              1, 1, 0, 0, 0, 0, 0, 0, 1.0f, 0, false);
  // 2) qf = question @ qst_fc_w + b
  launch_gemm(stream, question, qst_fc_w, qst_fc_b, QF, Bn * LQ, Dd, 300, 300, Dd, Dd,
              1, 1, 0, 0, 0, 0, 0, 0, 1.0f, 0, false);

  // 3) question self-attn (h=4) + qst_query_block -> QSTF
  run_mha(stream, QF, LQ, QF, LQ, attnq_in_w, attnq_in_b, 1536,
          attnq_out_w, attnq_out_b, 4, QPs, KPs, VPs, SCq, AOq, APq);
  run_qq_block(stream, APq, Bn * LQ, qq_lin1_w, qq_lin1_b, qq_lin2_w, qq_lin2_b,
               qq_ln, T1s, T2s, QSTF);

  // 4) s1 = cross-attn(af, qf) h=1 -> R1
  run_mha(stream, AF, T, QF, LQ, cm_in_w, cm_in_b, 1536,
          cm_out_w, cm_out_b, 1, G1, KPs, VPs, SC, O1, R1);
  // 5) s2 = self-attn(af) h=1 -> R2
  run_mha(stream, AF, T, AF, T, sa_in_w, sa_in_b, 1536,
          sa_out_w, sa_out_b, 1, G1, G2, G3, SC, O1, R2);
  // 6) x = LN(af + s1 + s2) -> G1
  ln512_kernel<<<Bn * T, 256, 0, stream>>>(AF, R1, R2, ln1, G1, 0);
  // 7) audio_feat = LN(x + relu(x@ff1)@ff2) -> R1
  launch_gemm(stream, G1, ff1_w, ff1_b, G2, Bn * T, Dd, Dd, Dd, Dd, Dd,
              1, 1, 0, 0, 0, 0, 0, 0, 1.0f, 1, false);
  launch_gemm(stream, G2, ff2_w, ff2_b, G3, Bn * T, Dd, Dd, Dd, Dd, Dd,
              1, 1, 0, 0, 0, 0, 0, 0, 1.0f, 0, false);
  ln512_kernel<<<Bn * T, 256, 0, stream>>>(G1, G3, nullptr, ln2, R1, 0);

  // 8) multi-window sliding attention, ms accumulated in R2
  hipMemsetAsync(R2, 0, big * sizeof(float), stream);
  const int windows[4] = {2, 4, 6, 12};
  const float sscale = 1.0f / sqrtf(128.0f);
  const int sblocks = (Bn * T * 4 + 7) / 8;
  for (int i = 0; i < 4; ++i) {
    const float* qw = ms_qkv_w + ((size_t)i * 3 + 0) * 512 * 512;
    const float* kw = ms_qkv_w + ((size_t)i * 3 + 1) * 512 * 512;
    const float* vw = ms_qkv_w + ((size_t)i * 3 + 2) * 512 * 512;
    const float* qb = ms_qkv_b + ((size_t)i * 3 + 0) * 512;
    const float* kb = ms_qkv_b + ((size_t)i * 3 + 1) * 512;
    const float* vb = ms_qkv_b + ((size_t)i * 3 + 2) * 512;
    launch_gemm(stream, R1, qw, qb, G1, Bn * T, Dd, Dd, Dd, Dd, Dd,
                1, 1, 0, 0, 0, 0, 0, 0, 1.0f, 0, false);
    launch_gemm(stream, R1, kw, kb, G2, Bn * T, Dd, Dd, Dd, Dd, Dd,
                1, 1, 0, 0, 0, 0, 0, 0, 1.0f, 0, false);
    launch_gemm(stream, R1, vw, vb, G3, Bn * T, Dd, Dd, Dd, Dd, Dd,
                1, 1, 0, 0, 0, 0, 0, 0, 1.0f, 0, false);
    switch (windows[i]) {
      case 2:  sliding_attn_kernel<2><<<sblocks, 256, 0, stream>>>(G1, G2, G3, O1, Bn, T, sscale); break;
      case 4:  sliding_attn_kernel<4><<<sblocks, 256, 0, stream>>>(G1, G2, G3, O1, Bn, T, sscale); break;
      case 6:  sliding_attn_kernel<6><<<sblocks, 256, 0, stream>>>(G1, G2, G3, O1, Bn, T, sscale); break;
      default: sliding_attn_kernel<12><<<sblocks, 256, 0, stream>>>(G1, G2, G3, O1, Bn, T, sscale); break;
    }
    ln512_kernel<<<Bn * T, 256, 0, stream>>>(R1, O1, nullptr, ms_ln + (size_t)i * 1024, R2, 1);
  }
  // 9) audio_feat2 = LN(audio_feat + 0.25*ms@msl_w + msl_b) -> G2
  launch_gemm(stream, R2, msl_w, msl_b, G1, Bn * T, Dd, Dd, Dd, Dd, Dd,
              1, 1, 0, 0, 0, 0, 0, 0, 0.25f, 0, false);
  ln512_kernel<<<Bn * T, 256, 0, stream>>>(R1, G1, nullptr, ms_norm, G2, 0);

  // 10) av = qq_block(mha(audio_feat2, qst_feat, qst_feat, attnq, h=4)) -> G1
  run_mha(stream, G2, T, QSTF, LQ, attnq_in_w, attnq_in_b, 1536,
          attnq_out_w, attnq_out_b, 4, G1, KPs, VPs, SC, O1, G3);
  run_qq_block(stream, G3, Bn * T, qq_lin1_w, qq_lin1_b, qq_lin2_w, qq_lin2_b,
               qq_ln, R1, R2, G1);

  // 11) QA layer i=3 on av (G1)
  const float* qaw  = qa_in_w  + (size_t)3 * 512 * 1536;
  const float* qab  = qa_in_b  + (size_t)3 * 1536;
  const float* qaow = qa_out_w + (size_t)3 * 512 * 512;
  const float* qaob = qa_out_b + (size_t)3 * 512;
  run_mha(stream, G1, T, G1, T, qaw, qab, 1536, qaow, qaob, 1, G2, G3, R1, SC, O1, R2);
  ln512_kernel<<<Bn * T, 256, 0, stream>>>(G1, R2, nullptr, qa_ln1 + (size_t)3 * 1024, G2, 0);
  launch_gemm(stream, G2, qa_ff1_w + (size_t)3 * 512 * 512, qa_ff1_b + (size_t)3 * 512,
              G3, Bn * T, Dd, Dd, Dd, Dd, Dd, 1, 1, 0, 0, 0, 0, 0, 0, 1.0f, 1, false);
  launch_gemm(stream, G3, qa_ff2_w + (size_t)3 * 512 * 512, qa_ff2_b + (size_t)3 * 512,
              R1, Bn * T, Dd, Dd, Dd, Dd, Dd, 1, 1, 0, 0, 0, 0, 0, 0, 1.0f, 0, false);
  ln512_kernel<<<Bn * T, 256, 0, stream>>>(G2, R1, nullptr, qa_ln2 + (size_t)3 * 1024, O1, 0);

  // 12) comb = [mean(fused), mean(qst_feat)]
  mean_axis1_kernel<<<(Bn * Dd + 255) / 256, 256, 0, stream>>>(O1, COMB, Bn, T, Dd, 1024, 0);
  mean_axis1_kernel<<<(Bn * Dd + 255) / 256, 256, 0, stream>>>(QSTF, COMB, Bn, LQ, Dd, 1024, 512);

  // 13-15) classifier head
  launch_gemm(stream, COMB, fc1_w, fc1_b, C1, Bn, 512, 1024, 1024, 512, 512,
              1, 1, 0, 0, 0, 0, 0, 0, 1.0f, 2, false);
  launch_gemm(stream, C1, fc2_w, fc2_b, C2, Bn, 256, 512, 512, 256, 256,
              1, 1, 0, 0, 0, 0, 0, 0, 1.0f, 2, false);
  launch_gemm(stream, C2, pred_w, pred_b, (float*)d_out, Bn, 828, 256, 256, 828, 828,
              1, 1, 0, 0, 0, 0, 0, 0, 1.0f, 0, false);
}
